// OPTDecoder_26345329393727
// MI455X (gfx1250) — matrix operand order repaired and checked by reading, not compiled, not run on hardware
//
#include <hip/hip_runtime.h>
#include <hip/hip_bf16.h>
#include <cstdint>
#include <cstddef>

// Problem constants (match reference)
#define V_  50272
#define DW_ 512
#define D_  1024
#define H_  16
#define F_  4096
#define NL_ 4
#define B_  4
#define L_  1024
#define T_  4096
#define HD_ 64

typedef __attribute__((ext_vector_type(16))) __bf16 v16bf;
typedef __attribute__((ext_vector_type(8)))  __bf16 v8bf;
typedef __attribute__((ext_vector_type(8)))  float  v8f;
typedef unsigned short u16;

// ---------- helpers ----------
__device__ __forceinline__ u16 f2bf(float f) {
  union { float f; unsigned u; } v; v.f = f;
  unsigned u = v.u;
  u += 0x7FFFu + ((u >> 16) & 1u);          // round-to-nearest-even
  return (u16)(u >> 16);
}

// A operand (16x32 MxK), source row-major [rows,K]. No bounds check: callers
// guarantee rows are valid (buffers padded to tile multiples).
__device__ __forceinline__ v16bf load_a_tile(const u16* base, int ld, int row0,
                                             int k0, int lane) {
  int r   = row0 + (lane & 15);
  int off = (lane >> 4) * 16;
  const v8bf* p0 = (const v8bf*)(base + (size_t)r * ld + k0 + off);
  const v8bf* p1 = (const v8bf*)(base + (size_t)r * ld + k0 + 8 + off);
  v8bf lo = *p0, hi = *p1;
  return __builtin_shufflevector(lo, hi, 0,1,2,3,4,5,6,7,8,9,10,11,12,13,14,15);
}

// B operand (32x16 KxN) from row-major [N,K] source.
__device__ __forceinline__ v16bf load_b_tile(const u16* base, int ld, int n0,
                                             int k0, int lane) {
  int n  = n0 + (lane & 15);
  int kh = (lane >> 4) * 16;
  const v8bf* p0 = (const v8bf*)(base + (size_t)n * ld + k0 + kh);
  v8bf lo = p0[0], hi = p0[1];
  return __builtin_shufflevector(lo, hi, 0,1,2,3,4,5,6,7,8,9,10,11,12,13,14,15);
}

// B operand from row-major [K,N] source (V matrix: rows are keys).
__device__ __forceinline__ v16bf load_bT_tile(const u16* base, int ld, int k0,
                                              int n0, int lane) {
  int n  = n0 + (lane & 15);
  int kh = (lane >> 4) * 16;
  union { v16bf v; u16 u[16]; } r;
#pragma unroll
  for (int i = 0; i < 16; ++i)
    r.u[i] = base[(size_t)(k0 + kh + i) * ld + n];
  return r.v;
}

// Async global->LDS copy (CDNA5): per-lane 8-byte transfer, tracked by ASYNCcnt.
__device__ __forceinline__ void async_copy_b64(unsigned lds_dst, const u16* gsrc) {
  asm volatile("global_load_async_to_lds_b64 %0, %1, off"
               :: "v"(lds_dst), "v"((unsigned long long)(size_t)gsrc)
               : "memory");
}
__device__ __forceinline__ void async_wait0() {
  asm volatile("s_wait_asynccnt 0" ::: "memory");
}
__device__ __forceinline__ unsigned lds_off_of(const void* p) {
  return (unsigned)(size_t)p;   // low 32 bits of a flat shared pointer = LDS offset
}

// ---------- elementwise / gather kernels ----------
__global__ __launch_bounds__(256)
void opt_f32_to_bf16(const float* __restrict__ in, u16* __restrict__ out, size_t n) {
  size_t i = (size_t)blockIdx.x * 256 + threadIdx.x;
  if (i < n) out[i] = f2bf(in[i]);
}

__global__ __launch_bounds__(256)
void opt_zero_u16(u16* __restrict__ p, size_t n) {
  size_t i = (size_t)blockIdx.x * 256 + threadIdx.x;
  if (i < n) p[i] = 0;
}

__global__ __launch_bounds__(256)
void opt_gather_bf16(const u16* __restrict__ table, const int* __restrict__ idx,
                     int add, u16* __restrict__ out, int rowlen) {
  int r = blockIdx.x;
  const u16* s = table + (size_t)(idx[r] + add) * rowlen;
  u16* d = out + (size_t)r * rowlen;
  for (int i = threadIdx.x; i < rowlen; i += 256) d[i] = s[i];
}

__global__ __launch_bounds__(256)
void opt_gather_f32(const float* __restrict__ table, const int* __restrict__ idx,
                    int add, float* __restrict__ out, int rowlen) {
  int r = blockIdx.x;
  const float* s = table + (size_t)(idx[r] + add) * rowlen;
  float* d = out + (size_t)r * rowlen;
  for (int i = threadIdx.x; i < rowlen; i += 256) d[i] = s[i];
}

// ---------- GEMM: C[M,N] = A[M,K] * B[N,K]^T (+bias +resid, ReLU) ----------
// Block (256 thr, 8 waves) computes 256(M) x 32(N). The 32x32 B chunk per
// k-step is staged into double-buffered LDS with async copies and shared by
// all 8 waves. Each wave computes a 32x32 tile: 4 WMMAs per k-chunk.
__device__ __forceinline__ void epilogue_tile(const v8f& acc, int m0, int n,
                                              float bv, const float* resid,
                                              float* Cf, u16* Cb,
                                              int M, int N, int relu, int mb) {
#pragma unroll
  for (int i = 0; i < 8; ++i) {
    int m = m0 + mb + i;
    if (m < M) {
      float v = acc[i] + bv;
      if (resid) v += resid[(size_t)m * N + n];
      if (relu)  v = fmaxf(v, 0.f);
      if (Cf) Cf[(size_t)m * N + n] = v;
      if (Cb) Cb[(size_t)m * N + n] = f2bf(v);
    }
  }
}

__global__ __launch_bounds__(256)
void opt_gemm_bf16(const u16* __restrict__ A, const u16* __restrict__ Bw,
                   const float* __restrict__ bias, const float* __restrict__ resid,
                   float* __restrict__ Cf, u16* __restrict__ Cb,
                   int M, int N, int K, int relu) {
  __shared__ u16 sB[2][32 * 32];

  int tid    = threadIdx.x;
  int lane   = tid & 31;
  int wave   = tid >> 5;
  int tilesN = N >> 5;
  int tm = (blockIdx.x / tilesN) << 8;   // 256-row block
  int tn = (blockIdx.x % tilesN) << 5;   // 32-col block
  int wm = tm + wave * 32;

  // Staging task: thread copies 8B of the 32x32 B chunk. row=tid>>3, 4-half seg.
  int srow = tid >> 3;
  int sseg = (tid & 7) * 4;
  const u16* gB = Bw + (size_t)(tn + srow) * K + sseg;
  unsigned ldsElem = (unsigned)(srow * 32 + sseg);

  async_copy_b64(lds_off_of(&sB[0][ldsElem]), gB);   // stage chunk 0

  v8f z = {0.f,0.f,0.f,0.f,0.f,0.f,0.f,0.f};
  v8f acc00 = z, acc01 = z, acc10 = z, acc11 = z;

  int nk = K >> 5;
  for (int kc = 0; kc < nk; ++kc) {
    async_wait0();
    __syncthreads();                      // buffer kc ready for all waves
    if (kc + 1 < nk)
      async_copy_b64(lds_off_of(&sB[(kc + 1) & 1][ldsElem]), gB + (kc + 1) * 32);

    const u16* bb = sB[kc & 1];
    int k = kc << 5;
    if (k + 32 < K)
      __builtin_prefetch(A + (size_t)(wm + (lane & 15)) * K + k + 32, 0, 1);

    v16bf a0 = load_a_tile(A, K, wm,      k, lane);
    v16bf a1 = load_a_tile(A, K, wm + 16, k, lane);
    v16bf b0 = load_b_tile(bb, 32, 0,  0, lane);
    v16bf b1 = load_b_tile(bb, 32, 16, 0, lane);
    acc00 = __builtin_amdgcn_wmma_f32_16x16x32_bf16(false, a0, false, b0, (short)0, acc00, false, false);
    acc01 = __builtin_amdgcn_wmma_f32_16x16x32_bf16(false, a0, false, b1, (short)0, acc01, false, false);
    acc10 = __builtin_amdgcn_wmma_f32_16x16x32_bf16(false, a1, false, b0, (short)0, acc10, false, false);
    acc11 = __builtin_amdgcn_wmma_f32_16x16x32_bf16(false, a1, false, b1, (short)0, acc11, false, false);
    __syncthreads();                      // all reads of buffer done before reuse
  }

  int nn = tn + (lane & 15);
  int mb = (lane >> 4) << 3;
  float bv0 = bias ? bias[nn]      : 0.f;
  float bv1 = bias ? bias[nn + 16] : 0.f;
  epilogue_tile(acc00, wm,      nn,      bv0, resid, Cf, Cb, M, N, relu, mb);
  epilogue_tile(acc01, wm,      nn + 16, bv1, resid, Cf, Cb, M, N, relu, mb);
  epilogue_tile(acc10, wm + 16, nn,      bv0, resid, Cf, Cb, M, N, relu, mb);
  epilogue_tile(acc11, wm + 16, nn + 16, bv1, resid, Cf, Cb, M, N, relu, mb);
}

// ---------- Flash attention: one wave per (batch, head, 16-query tile) ----------
__global__ __launch_bounds__(32)
void opt_flash_attn(const u16* __restrict__ Q, const u16* __restrict__ Kc,
                    const u16* __restrict__ Vc, u16* __restrict__ O) {
  __shared__ float sS[16 * 32];
  __shared__ u16   sP[16 * 32];
  __shared__ float sScale[16];

  int lane = threadIdx.x;
  int bid  = blockIdx.x;
  int qt = bid & 63;           // L/16 = 64 query tiles
  int h  = (bid >> 6) & 15;
  int b  = bid >> 10;
  int c0   = h * HD_;
  int rowq = b * L_ + qt * 16;
  const float scale = 0.125f;  // hd^-0.5

  v16bf qa0 = load_a_tile(Q, D_, rowq, c0,      lane);
  v16bf qa1 = load_a_tile(Q, D_, rowq, c0 + 32, lane);

  v8f zr = {0.f,0.f,0.f,0.f,0.f,0.f,0.f,0.f};
  v8f oacc[4];
#pragma unroll
  for (int t = 0; t < 4; ++t) oacc[t] = zr;

  float rowM = -3.0e38f, rowL = 0.f;
  int nlane = lane & 15;
  int mbase = (lane >> 4) << 3;
  int nchunks = (qt * 16 + 15) / 32 + 1;   // causal chunk skipping

  for (int kc = 0; kc < nchunks; ++kc) {
    int key0 = kc * 32;
#pragma unroll
    for (int j = 0; j < 2; ++j) {
      v16bf kb0 = load_b_tile(Kc, D_, b * L_ + key0 + j * 16, c0,      lane);
      v16bf kb1 = load_b_tile(Kc, D_, b * L_ + key0 + j * 16, c0 + 32, lane);
      v8f s = zr;
      s = __builtin_amdgcn_wmma_f32_16x16x32_bf16(false, qa0, false, kb0, (short)0, s, false, false);
      s = __builtin_amdgcn_wmma_f32_16x16x32_bf16(false, qa1, false, kb1, (short)0, s, false, false);
#pragma unroll
      for (int i = 0; i < 8; ++i) {
        int m  = mbase + i;
        int gk = key0 + j * 16 + nlane;
        int gq = qt * 16 + m;
        float sv = s[i] * scale;
        if (gk > gq) sv = -3.0e38f;
        sS[m * 32 + j * 16 + nlane] = sv;
      }
    }
    __syncthreads();
    if (lane < 16) {          // online softmax: lane owns one query row
      float cmax = -3.0e38f;
#pragma unroll 4
      for (int c = 0; c < 32; ++c) cmax = fmaxf(cmax, sS[lane * 32 + c]);
      float Mn = fmaxf(rowM, cmax);
      float f  = __expf(rowM - Mn);
      float sum = 0.f;
#pragma unroll 4
      for (int c = 0; c < 32; ++c) {
        float p = __expf(sS[lane * 32 + c] - Mn);
        sP[lane * 32 + c] = f2bf(p);
        sum += p;
      }
      rowL = rowL * f + sum;
      rowM = Mn;
      sScale[lane] = f;
    }
    __syncthreads();
#pragma unroll
    for (int i = 0; i < 8; ++i) {
      float f = sScale[mbase + i];
#pragma unroll
      for (int t = 0; t < 4; ++t) oacc[t][i] *= f;
    }
    v16bf pa = load_a_tile(sP, 32, 0, 0, lane);
#pragma unroll
    for (int t = 0; t < 4; ++t) {
      v16bf vbt = load_bT_tile(Vc, D_, b * L_ + key0, c0 + t * 16, lane);
      oacc[t] = __builtin_amdgcn_wmma_f32_16x16x32_bf16(false, pa, false, vbt, (short)0, oacc[t], false, false);
    }
    __syncthreads();
  }
  if (lane < 16) sScale[lane] = 1.f / rowL;
  __syncthreads();
#pragma unroll
  for (int i = 0; i < 8; ++i) {
    int m = mbase + i;
    float inv = sScale[m];
    size_t base = (size_t)(rowq + m) * D_ + c0 + nlane;
    O[base +  0] = f2bf(oacc[0][i] * inv);
    O[base + 16] = f2bf(oacc[1][i] * inv);
    O[base + 32] = f2bf(oacc[2][i] * inv);
    O[base + 48] = f2bf(oacc[3][i] * inv);
  }
}

// ---------- LayerNorm ----------
__global__ __launch_bounds__(256)
void opt_layernorm(const float* __restrict__ x, const float* __restrict__ w,
                   const float* __restrict__ b, float* __restrict__ outf,
                   u16* __restrict__ outb, int n) {
  __shared__ float red[256];
  int row = blockIdx.x;
  const float* xr = x + (size_t)row * n;
  float s = 0.f;
  for (int i = threadIdx.x; i < n; i += 256) s += xr[i];
  red[threadIdx.x] = s; __syncthreads();
  for (int o2 = 128; o2 > 0; o2 >>= 1) {
    if (threadIdx.x < (unsigned)o2) red[threadIdx.x] += red[threadIdx.x + o2];
    __syncthreads();
  }
  float mean = red[0] / n;
  __syncthreads();
  float v = 0.f;
  for (int i = threadIdx.x; i < n; i += 256) { float d = xr[i] - mean; v += d * d; }
  red[threadIdx.x] = v; __syncthreads();
  for (int o2 = 128; o2 > 0; o2 >>= 1) {
    if (threadIdx.x < (unsigned)o2) red[threadIdx.x] += red[threadIdx.x + o2];
    __syncthreads();
  }
  float inv = rsqrtf(red[0] / n + 1e-5f);
  for (int i = threadIdx.x; i < n; i += 256) {
    float y = (xr[i] - mean) * inv * w[i] + b[i];
    outf[(size_t)row * n + i] = y;
    outb[(size_t)row * n + i] = f2bf(y);
  }
}

// ---------- host orchestration ----------
extern "C" void kernel_launch(void* const* d_in, const int* in_sizes, int n_in,
                              void* d_out, int out_size, void* d_ws, size_t ws_size,
                              hipStream_t stream) {
  (void)in_sizes; (void)n_in; (void)out_size; (void)ws_size;
  const int*   input_ids  = (const int*)  d_in[0];
  const int*   positions  = (const int*)  d_in[1];
  const int*   offsets    = (const int*)  d_in[2];
  const float* embed_w    = (const float*)d_in[6];
  const float* pos_w      = (const float*)d_in[7];
  const float* proj_in_w  = (const float*)d_in[8];
  const float* proj_out_w = (const float*)d_in[9];
  const float* ln1_w = (const float*)d_in[10];
  const float* ln1_b = (const float*)d_in[11];
  const float* qw = (const float*)d_in[12];
  const float* qb = (const float*)d_in[13];
  const float* kw = (const float*)d_in[14];
  const float* kb = (const float*)d_in[15];
  const float* vw = (const float*)d_in[16];
  const float* vb = (const float*)d_in[17];
  const float* ow = (const float*)d_in[18];
  const float* ob = (const float*)d_in[19];
  const float* fc1w = (const float*)d_in[20];
  const float* fc1b = (const float*)d_in[21];
  const float* fc2w = (const float*)d_in[22];
  const float* fc2b = (const float*)d_in[23];
  const float* ln2_w = (const float*)d_in[24];
  const float* ln2_b = (const float*)d_in[25];
  float* logits = (float*)d_out;

  size_t off = 0;
  auto alloc = [&](size_t bytes) -> char* {
    size_t p = (off + 255) & ~(size_t)255;
    off = p + bytes;
    return (char*)d_ws + p;
  };
  u16*   embBF  = (u16*)  alloc((size_t)V_ * DW_ * 2);
  u16*   pinBF  = (u16*)  alloc((size_t)D_ * DW_ * 2);
  u16*   poutBF = (u16*)  alloc((size_t)DW_ * D_ * 2);
  u16*   qwBF   = (u16*)  alloc((size_t)NL_ * D_ * D_ * 2);
  u16*   kwBF   = (u16*)  alloc((size_t)NL_ * D_ * D_ * 2);
  u16*   vwBF   = (u16*)  alloc((size_t)NL_ * D_ * D_ * 2);
  u16*   owBF   = (u16*)  alloc((size_t)NL_ * D_ * D_ * 2);
  u16*   fc1BF  = (u16*)  alloc((size_t)NL_ * F_ * D_ * 2);
  u16*   fc2BF  = (u16*)  alloc((size_t)NL_ * D_ * F_ * 2);
  u16*   embx   = (u16*)  alloc((size_t)T_ * DW_ * 2);
  float* posf   = (float*)alloc((size_t)T_ * D_ * 4);
  float* hf     = (float*)alloc((size_t)T_ * D_ * 4);
  u16*   hb     = (u16*)  alloc((size_t)T_ * D_ * 2);
  u16*   qbuf   = (u16*)  alloc((size_t)T_ * D_ * 2);
  u16*   kbuf   = (u16*)  alloc((size_t)T_ * D_ * 2);
  u16*   vbuf   = (u16*)  alloc((size_t)T_ * D_ * 2);
  u16*   obuf   = (u16*)  alloc((size_t)T_ * D_ * 2);
  float* tmpf   = (float*)alloc((size_t)T_ * D_ * 4);
  u16*   ffnb   = (u16*)  alloc((size_t)T_ * F_ * 2);
  u16*   yb     = (u16*)  alloc((size_t)T_ * DW_ * 2);
  u16*   lastb  = (u16*)  alloc((size_t)256 * DW_ * 2);   // padded to one M-block

  auto cvt = [&](const float* src, u16* dst, size_t n) {
    int blocks = (int)((n + 255) / 256);
    opt_f32_to_bf16<<<blocks, 256, 0, stream>>>(src, dst, n);
  };
  cvt(embed_w,    embBF,  (size_t)V_ * DW_);
  cvt(proj_in_w,  pinBF,  (size_t)D_ * DW_);
  cvt(proj_out_w, poutBF, (size_t)DW_ * D_);
  cvt(qw,   qwBF,  (size_t)NL_ * D_ * D_);
  cvt(kw,   kwBF,  (size_t)NL_ * D_ * D_);
  cvt(vw,   vwBF,  (size_t)NL_ * D_ * D_);
  cvt(ow,   owBF,  (size_t)NL_ * D_ * D_);
  cvt(fc1w, fc1BF, (size_t)NL_ * F_ * D_);
  cvt(fc2w, fc2BF, (size_t)NL_ * D_ * F_);

  opt_gather_bf16<<<T_, 256, 0, stream>>>(embBF, input_ids, 0, embx, DW_);
  opt_gather_f32 <<<T_, 256, 0, stream>>>(pos_w, positions, 2, posf, D_);

  auto gemm = [&](const u16* A, const u16* Bw, const float* bias, const float* resid,
                  float* Cf, u16* Cb, int M, int N, int K, int relu) {
    int tilesM = (M + 255) / 256;
    int tilesN = N / 32;
    opt_gemm_bf16<<<tilesM * tilesN, 256, 0, stream>>>(A, Bw, bias, resid, Cf, Cb,
                                                       M, N, K, relu);
  };

  // h0 = embed @ proj_in^T + pos_emb  -> hf (f32) + hb (bf16)
  gemm(embx, pinBF, nullptr, posf, hf, hb, T_, D_, DW_, 0);

  for (int l = 0; l < NL_; ++l) {
    const size_t wDD = (size_t)D_ * D_;
    gemm(hb, qwBF + l * wDD, qb + (size_t)l * D_, nullptr, nullptr, qbuf, T_, D_, D_, 0);
    gemm(hb, kwBF + l * wDD, kb + (size_t)l * D_, nullptr, nullptr, kbuf, T_, D_, D_, 0);
    gemm(hb, vwBF + l * wDD, vb + (size_t)l * D_, nullptr, nullptr, vbuf, T_, D_, D_, 0);
    opt_flash_attn<<<B_ * H_ * (L_ / 16), 32, 0, stream>>>(qbuf, kbuf, vbuf, obuf);
    gemm(obuf, owBF + l * wDD, ob + (size_t)l * D_, hf, tmpf, nullptr, T_, D_, D_, 0);
    opt_layernorm<<<T_, 256, 0, stream>>>(tmpf, ln1_w + (size_t)l * D_,
                                          ln1_b + (size_t)l * D_, hf, hb, D_);
    gemm(hb, fc1BF + (size_t)l * F_ * D_, fc1b + (size_t)l * F_, nullptr, nullptr,
         ffnb, T_, F_, D_, 1);
    gemm(ffnb, fc2BF + (size_t)l * D_ * F_, fc2b + (size_t)l * D_, hf, tmpf, nullptr,
         T_, D_, F_, 0);
    opt_layernorm<<<T_, 256, 0, stream>>>(tmpf, ln2_w + (size_t)l * D_,
                                          ln2_b + (size_t)l * D_, hf, hb, D_);
  }

  // y = h @ proj_out^T ; last tokens (zero-padded to 256 rows) ; logits
  gemm(hb, poutBF, nullptr, nullptr, nullptr, yb, T_, DW_, D_, 0);
  opt_zero_u16<<<(256 * DW_ + 255) / 256, 256, 0, stream>>>(lastb, (size_t)256 * DW_);
  opt_gather_bf16<<<B_, 256, 0, stream>>>(yb, offsets, -1, lastb, DW_);
  gemm(lastb, embBF, nullptr, nullptr, logits, nullptr, B_, V_, DW_, 0);
}
